// GRUD_75376676045474
// MI455X (gfx1250) — matrix operand order repaired and checked by reading, not compiled, not run on hardware
//
#include <hip/hip_runtime.h>

typedef __attribute__((ext_vector_type(16))) __bf16 v16bf;
typedef __attribute__((ext_vector_type(8)))  float  v8f;
typedef __attribute__((ext_vector_type(4)))  int    v4i;

constexpr int NB = 256;   // batch
constexpr int NT = 256;   // time
constexpr int ND = 128;   // feature dim
constexpr int NH = 256;   // hidden
constexpr int BT = NB * NT;

// ---------------- CDNA5 async global->LDS copy (ASYNCcnt-tracked) ----------------
#if __has_builtin(__builtin_amdgcn_global_load_async_to_lds_b128)
#define HAVE_ASYNC 1
#else
#define HAVE_ASYNC 0
#endif

typedef __attribute__((address_space(1))) v4i* gv4_t;   // global 16B chunk
typedef __attribute__((address_space(3))) v4i* lv4_t;   // LDS 16B chunk

static __device__ inline void async_cp16(const void* g, void* l) {
#if HAVE_ASYNC
  __builtin_amdgcn_global_load_async_to_lds_b128((gv4_t)const_cast<void*>(g),
                                                 (lv4_t)l, 0, 0);
#else
  (void)g; (void)l;
#endif
}

static __device__ inline void wait_async_le16() {
#if HAVE_ASYNC
#if __has_builtin(__builtin_amdgcn_s_wait_asynccnt)
  __builtin_amdgcn_s_wait_asynccnt(16);
#else
  asm volatile("s_wait_asynccnt 0x10" ::: "memory");
#endif
#endif
}
static __device__ inline void wait_async_le0() {
#if HAVE_ASYNC
#if __has_builtin(__builtin_amdgcn_s_wait_asynccnt)
  __builtin_amdgcn_s_wait_asynccnt(0);
#else
  asm volatile("s_wait_asynccnt 0x0" ::: "memory");
#endif
#endif
}

// ---------------- WMMA fragment helpers (CDNA5 16x16x32 bf16) ----------------
// A fragment: 16x32 (MxK), 16-bit. Lane l: m = l&15, K-band = (l>>4)*8.
// elements 0..7 -> K = band+0..7 ; elements 8..15 -> K = band+16..23.
static __device__ inline v16bf load_a16(const __bf16* base, int lda, int lane) {
  int m  = lane & 15;
  int kb = (lane >> 4) << 3;
  const __bf16* p = base + (size_t)m * lda + kb;
  v16bf a;
#pragma unroll
  for (int i = 0; i < 8; ++i) { a[i] = p[i]; a[i + 8] = p[i + 16]; }
  return a;
}

// B fragment: 32x16 (KxN), 16-bit, stored [n][k]. Lane l: n = l&15,
// K-band = (l>>4)*8; elements 0..7 -> K = band+0..7 ; elements 8..15 -> K = band+16..23.
static __device__ inline v16bf load_b16(const __bf16* base, int ldb, int lane) {
  int n  = lane & 15;
  int ko = (lane >> 4) << 3;
  const __bf16* p = base + (size_t)n * ldb + ko;
  v16bf b;
#pragma unroll
  for (int i = 0; i < 8; ++i) { b[i] = p[i]; b[i + 8] = p[i + 16]; }
  return b;
}

static __device__ inline v8f wmma_bf16(v16bf a, v16bf b, v8f c) {
  return __builtin_amdgcn_wmma_f32_16x16x32_bf16(
      /*neg_a=*/false, a, /*neg_b=*/false, b,
      /*c_mod=*/(short)0, c, /*reuse_a=*/false, /*reuse_b=*/false);
}

// ---------------- 1) weight slicing / bf16 conversion ----------------
__global__ void prep_weights(const float* __restrict__ Wz, const float* __restrict__ Wr,
                             const float* __restrict__ Wh, const float* __restrict__ Wgh,
                             const float* __restrict__ bz, const float* __restrict__ br,
                             __bf16* dWzrH, __bf16* dWzrXM, __bf16* dWhH,
                             __bf16* dWhXM, __bf16* dWgh, float* dBiasZr) {
  int idx = blockIdx.x * blockDim.x + threadIdx.x;
  if (idx < 131072) {                       // WzrH (512 x 256), h-part cols 128..383
    int n = idx >> 8, k = idx & 255;
    const float* W = (n < 256) ? Wz : Wr; int nn = n & 255;
    dWzrH[idx] = (__bf16)W[nn * 512 + 128 + k];
  } else if (idx < 262144) {                // WzrXM (512 x 256), x|m-part
    int l = idx - 131072; int n = l >> 8, k = l & 255;
    const float* W = (n < 256) ? Wz : Wr; int nn = n & 255;
    int c = (k < 128) ? k : 384 + (k - 128);
    dWzrXM[l] = (__bf16)W[nn * 512 + c];
  } else if (idx < 327680) {                // WhH (256 x 256)
    int l = idx - 262144; int n = l >> 8, k = l & 255;
    dWhH[l] = (__bf16)Wh[n * 512 + 128 + k];
  } else if (idx < 393216) {                // WhXM (256 x 256)
    int l = idx - 327680; int n = l >> 8, k = l & 255;
    int c = (k < 128) ? k : 384 + (k - 128);
    dWhXM[l] = (__bf16)Wh[n * 512 + c];
  } else if (idx < 425984) {                // Wgh (256 x 128)
    int l = idx - 393216;
    dWgh[l] = (__bf16)Wgh[l];
  } else if (idx < 426496) {                // bias_zr = [b_z | b_r]
    int l = idx - 425984;
    dBiasZr[l] = (l < 256) ? bz[l] : br[l - 256];
  }
}

// ---------------- 2) elementwise input prep ----------------
__global__ void build_inputs(const float* __restrict__ inp, const float* __restrict__ Xmean,
                             const float* __restrict__ Wgx, const float* __restrict__ bgx,
                             __bf16* __restrict__ xm, __bf16* __restrict__ dl) {
  int idx = blockIdx.x * blockDim.x + threadIdx.x;
  if (idx >= BT * ND) return;
  int d  = idx & (ND - 1);
  int bt = idx >> 7;
  int t  = bt & (NT - 1);
  int b  = bt >> 8;
  size_t base = (((size_t)b * 4 + 0) * NT + t) * ND + d;  // channel stride NT*ND
  float x  = inp[base];
  float xl = inp[base + (size_t)1 * NT * ND];
  float m  = inp[base + (size_t)2 * NT * ND];
  float dt = inp[base + (size_t)3 * NT * ND];
  if (x != x) x = -1.0f;  // nan_to_num(nan=-1)
  float dx = expf(-fmaxf(dt * Wgx[d * ND + d] + bgx[d], 0.0f));
  float xe = m * x + (1.0f - m) * (dx * xl + (1.0f - dx) * Xmean[t * ND + d]);
  xm[(size_t)bt * 256 + d]       = (__bf16)xe;
  xm[(size_t)bt * 256 + 128 + d] = (__bf16)m;
  dl[(size_t)bt * 128 + d]       = (__bf16)dt;
}

// ---------------- 3) big parallel GEMM: Out = A @ Bw^T(+bias) ----------------
// A: (M x K) bf16 row-major.  Bw: (N x K) bf16 [n][k].  Out: (M x N) f32.
// mode 0: Out = acc + bias[n] ; mode 1: Out = exp(-relu(acc + bias[n])).
__global__ __launch_bounds__(256) void gemm_pre(const __bf16* __restrict__ A,
                                                const __bf16* __restrict__ Bw,
                                                const float* __restrict__ bias,
                                                float* __restrict__ Out,
                                                int M, int N, int K, int mode) {
  int lane = threadIdx.x & 31;
  int tile = blockIdx.x * 8 + (threadIdx.x >> 5);
  int ntg  = N >> 6;
  int mt = tile / ntg, ng = tile % ntg;
  if (mt >= (M >> 4)) return;
  const __bf16* Arow = A + (size_t)(mt << 4) * K;
  v8f acc[4];
#pragma unroll
  for (int i = 0; i < 4; ++i)
#pragma unroll
    for (int j = 0; j < 8; ++j) acc[i][j] = 0.0f;
  int nk = K >> 5;
  for (int kt = 0; kt < nk; ++kt) {
    v16bf a = load_a16(Arow + kt * 32, K, lane);
    v16bf bfr[4];
#pragma unroll
    for (int i = 0; i < 4; ++i)
      bfr[i] = load_b16(Bw + (size_t)((ng << 6) + (i << 4)) * K + kt * 32, K, lane);
#pragma unroll
    for (int i = 0; i < 4; ++i) acc[i] = wmma_bf16(a, bfr[i], acc[i]);
  }
  int half = lane >> 4, nl = lane & 15;
#pragma unroll
  for (int i = 0; i < 4; ++i)
#pragma unroll
    for (int j = 0; j < 8; ++j) {
      int m = (mt << 4) + j + (half << 3);
      int n = (ng << 6) + (i << 4) + nl;
      float v = acc[i][j] + bias[n];
      if (mode) v = expf(-fmaxf(v, 0.0f));
      Out[(size_t)m * N + n] = v;
    }
}

// ---------------- 4) serial scan: 16 batch rows per workgroup ----------------
// Per step: h *= delta_h ; zr = sigmoid(pre_zr + h@WzrH^T) ;
// h_tilde = tanh(pre_h + (r*h)@WhH^T) ; h = (1-z)*h + z*h_tilde.
// Next step's pre-activations (64KB/WG) are double-buffered into LDS with
// GLOBAL_LOAD_ASYNC_TO_LDS_B128 while this step's WMMAs run.
__global__ __launch_bounds__(256) void scan_kernel(const float* __restrict__ pre_zr,
                                                   const float* __restrict__ pre_h,
                                                   const float* __restrict__ dlt_h,
                                                   const __bf16* __restrict__ WzrH,
                                                   const __bf16* __restrict__ WhH,
                                                   const float* __restrict__ Wout,
                                                   const float* __restrict__ bout,
                                                   float* __restrict__ out) {
  __shared__ float  h32[16][256];
  __shared__ float  zsh[16][256];
  __shared__ __bf16 hbf[16][256];
  __shared__ __bf16 rhbf[16][256];
  extern __shared__ char dynsm[];                      // 128 KB double buffer
  float* pzr_s = (float*)dynsm;                        // [2][16*512]
  float* ph_s  = (float*)(dynsm + 2 * 8192 * sizeof(float));   // [2][16*256]
  float* pdh_s = (float*)(dynsm + 2 * 12288 * sizeof(float));  // [2][16*256]

  int tid = threadIdx.x, lane = tid & 31, wv = tid >> 5;
  int b0 = blockIdx.x * 16;
  for (int i = tid; i < 4096; i += 256) (&h32[0][0])[i] = 0.0f;

#if HAVE_ASYNC
  auto prefetch_step = [&](int t) {
    int slot = t & 1;
    // pre_zr tile: 16 rows x 512 f32 -> 2048 x 16B ops (8/thread)
    for (int o = tid; o < 2048; o += 256) {
      int f = o << 2; int r = f >> 9, c = f & 511;
      async_cp16(pre_zr + ((size_t)(b0 + r) * NT + t) * 512 + c,
                 pzr_s + slot * 8192 + f);
    }
    // pre_h + delta_h tiles: 16 rows x 256 f32 each (4+4/thread)
    for (int o = tid; o < 1024; o += 256) {
      int f = o << 2; int r = f >> 8, c = f & 255;
      async_cp16(pre_h + ((size_t)(b0 + r) * NT + t) * 256 + c,
                 ph_s + slot * 4096 + f);
      async_cp16(dlt_h + ((size_t)(b0 + r) * NT + t) * 256 + c,
                 pdh_s + slot * 4096 + f);
    }
  };
  prefetch_step(0);
#endif

  int half = lane >> 4, nl = lane & 15;
#pragma unroll 1
  for (int t = 0; t < NT; ++t) {
    int slot = t & 1;
#if HAVE_ASYNC
    if (t + 1 < NT) { prefetch_step(t + 1); wait_async_le16(); }
    else           { wait_async_le0(); }
#endif
    __syncthreads();  // async data for step t visible; h32 updates from t-1 visible
    // decay h and refresh bf16 copy
#pragma unroll
    for (int i = 0; i < 16; ++i) {
      int e = tid + (i << 8); int r = e >> 8, c = e & 255;
#if HAVE_ASYNC
      float dh = pdh_s[slot * 4096 + r * 256 + c];
#else
      float dh = dlt_h[((size_t)(b0 + r) * NT + t) * NH + c];
#endif
      float hv = h32[r][c] * dh;
      h32[r][c] = hv;
      hbf[r][c] = (__bf16)hv;
    }
    __syncthreads();
    // z|r GEMM: wave wv covers zr columns [wv*64, wv*64+64)
    v8f acc[4];
#pragma unroll
    for (int i = 0; i < 4; ++i)
#pragma unroll
      for (int j = 0; j < 8; ++j) acc[i][j] = 0.0f;
#pragma unroll
    for (int kt = 0; kt < 8; ++kt) {
      v16bf a = load_a16(&hbf[0][0] + kt * 32, 256, lane);
      v16bf bfr[4];
#pragma unroll
      for (int i = 0; i < 4; ++i)
        bfr[i] = load_b16(WzrH + (size_t)((wv << 6) + (i << 4)) * 256 + kt * 32, 256, lane);
#pragma unroll
      for (int i = 0; i < 4; ++i) acc[i] = wmma_bf16(a, bfr[i], acc[i]);
    }
#pragma unroll
    for (int i = 0; i < 4; ++i)
#pragma unroll
      for (int j = 0; j < 8; ++j) {
        int m  = j + (half << 3);
        int zc = (wv << 6) + (i << 4) + nl;
#if HAVE_ASYNC
        float v = acc[i][j] + pzr_s[slot * 8192 + m * 512 + zc];
#else
        float v = acc[i][j] + pre_zr[((size_t)(b0 + m) * NT + t) * 512 + zc];
#endif
        float s = 1.0f / (1.0f + expf(-v));
        if (zc < 256) zsh[m][zc] = s;                            // z
        else rhbf[m][zc - 256] = (__bf16)(s * h32[m][zc - 256]); // r*h
      }
    __syncthreads();
    // h_tilde GEMM: wave wv covers columns [wv*32, wv*32+32)
    v8f hacc[2];
#pragma unroll
    for (int i = 0; i < 2; ++i)
#pragma unroll
      for (int j = 0; j < 8; ++j) hacc[i][j] = 0.0f;
#pragma unroll
    for (int kt = 0; kt < 8; ++kt) {
      v16bf a = load_a16(&rhbf[0][0] + kt * 32, 256, lane);
      v16bf bfr[2];
#pragma unroll
      for (int i = 0; i < 2; ++i)
        bfr[i] = load_b16(WhH + (size_t)((wv << 5) + (i << 4)) * 256 + kt * 32, 256, lane);
#pragma unroll
      for (int i = 0; i < 2; ++i) hacc[i] = wmma_bf16(a, bfr[i], hacc[i]);
    }
#pragma unroll
    for (int i = 0; i < 2; ++i)
#pragma unroll
      for (int j = 0; j < 8; ++j) {
        int m = j + (half << 3);
        int c = (wv << 5) + (i << 4) + nl;
#if HAVE_ASYNC
        float ht = tanhf(hacc[i][j] + ph_s[slot * 4096 + m * 256 + c]);
#else
        float ht = tanhf(hacc[i][j] + pre_h[((size_t)(b0 + m) * NT + t) * NH + c]);
#endif
        float zv = zsh[m][c];
        float ho = h32[m][c];
        h32[m][c] = (1.0f - zv) * ho + zv * ht;
      }
  }
  __syncthreads();
  // final projection: out[b] = h @ W_out^T + b_out
  if (tid < 16) {
    float s = 0.0f;
    for (int k = 0; k < NH; ++k) s += h32[tid][k] * Wout[k];
    out[b0 + tid] = s + bout[0];
  }
}

// ---------------- host launch ----------------
extern "C" void kernel_launch(void* const* d_in, const int* in_sizes, int n_in,
                              void* d_out, int out_size, void* d_ws, size_t ws_size,
                              hipStream_t stream) {
  (void)in_sizes; (void)n_in; (void)out_size; (void)ws_size;
  const float* inp   = (const float*)d_in[0];
  const float* Xmean = (const float*)d_in[2];
  const float* Wz    = (const float*)d_in[3];
  const float* bz    = (const float*)d_in[4];
  const float* Wr    = (const float*)d_in[5];
  const float* br    = (const float*)d_in[6];
  const float* Wh    = (const float*)d_in[7];
  const float* bh    = (const float*)d_in[8];
  const float* Wgx   = (const float*)d_in[9];
  const float* bgx   = (const float*)d_in[10];
  const float* Wgh   = (const float*)d_in[11];
  const float* bgh   = (const float*)d_in[12];
  const float* Wout  = (const float*)d_in[13];
  const float* bout  = (const float*)d_in[14];
  float* out = (float*)d_out;

  char* ws = (char*)d_ws;
  size_t off = 0;
  auto take = [&](size_t bytes) { void* p = ws + off; off += (bytes + 255) & ~(size_t)255; return p; };
  __bf16* xm     = (__bf16*)take((size_t)BT * 256 * sizeof(__bf16));
  __bf16* dl     = (__bf16*)take((size_t)BT * 128 * sizeof(__bf16));
  float*  preZr  = (float*) take((size_t)BT * 512 * sizeof(float));
  float*  preH   = (float*) take((size_t)BT * 256 * sizeof(float));
  float*  dltH   = (float*) take((size_t)BT * 256 * sizeof(float));
  __bf16* dWzrH  = (__bf16*)take((size_t)512 * 256 * sizeof(__bf16));
  __bf16* dWzrXM = (__bf16*)take((size_t)512 * 256 * sizeof(__bf16));
  __bf16* dWhH   = (__bf16*)take((size_t)256 * 256 * sizeof(__bf16));
  __bf16* dWhXM  = (__bf16*)take((size_t)256 * 256 * sizeof(__bf16));
  __bf16* dWgh   = (__bf16*)take((size_t)256 * 128 * sizeof(__bf16));
  float*  biasZr = (float*) take((size_t)512 * sizeof(float));

  prep_weights<<<(426496 + 255) / 256, 256, 0, stream>>>(Wz, Wr, Wh, Wgh, bz, br,
                                                         dWzrH, dWzrXM, dWhH, dWhXM, dWgh, biasZr);
  build_inputs<<<(BT * ND + 255) / 256, 256, 0, stream>>>(inp, Xmean, Wgx, bgx, xm, dl);

  // pre_zr = [x|m] @ [Wz|Wr]_xm^T + [bz|br]         (M=65536, N=512, K=256)
  gemm_pre<<<(BT / 16) * (512 / 64) / 8, 256, 0, stream>>>(xm, dWzrXM, biasZr, preZr, BT, 512, 256, 0);
  // pre_h  = [x|m] @ Wh_xm^T + bh                   (M=65536, N=256, K=256)
  gemm_pre<<<(BT / 16) * (256 / 64) / 8, 256, 0, stream>>>(xm, dWhXM, bh, preH, BT, 256, 256, 0);
  // delta_h = exp(-relu(Delta @ Wgh^T + bgh))        (M=65536, N=256, K=128)
  gemm_pre<<<(BT / 16) * (256 / 64) / 8, 256, 0, stream>>>(dl, dWgh, bgh, dltH, BT, 256, 128, 1);

  // 128 KB dynamic LDS for the double-buffered per-step pre-activations
  scan_kernel<<<NB / 16, 256, 131072, stream>>>(preZr, preH, dltH, dWzrH, dWhH, Wout, bout, out);
}
